// Llama4VisionAttention_89610197663869
// MI455X (gfx1250) — matrix operand order repaired and checked by reading, not compiled, not run on hardware
//
#include <hip/hip_runtime.h>

// ---------------- problem constants ----------------
#define BB     8
#define SS     1025
#define EE     1408
#define HH     16
#define DH     88          // head dim
#define DP     96          // padded head dim (3 x 32)
#define SPAD   1056        // padded seq (33 x 32 = 66 x 16)
#define MROWS  (BB*SS)     // 8200 rows for the projections
#define MPAD   8256        // 129 x 64, padded row count (pad rows zeroed)
#define PBELEM ((size_t)BB*HH*SPAD*DP)   // elems per padded bf16 Q/K/V buffer
#define XBELEM ((size_t)MPAD*EE)         // elems of padded bf16 activation buffer
#define WELEM  ((size_t)EE*EE)           // elems of one bf16 weight buffer

typedef __attribute__((ext_vector_type(16))) __bf16 v16bf;
typedef __attribute__((ext_vector_type(8)))  float  v8f;

union V16 { unsigned u[8]; v16bf v; };

static __device__ inline unsigned short f32_to_bf16(float f) {
  union { float f; unsigned u; } x; x.f = f;
  unsigned r = x.u + 0x7FFFu + ((x.u >> 16) & 1u);
  return (unsigned short)(r >> 16);
}
static __device__ inline unsigned pack_bf16x2(float lo, float hi) {
  return (unsigned)f32_to_bf16(lo) | ((unsigned)f32_to_bf16(hi) << 16);
}
static __device__ inline float bf16_to_f32(unsigned hs) {
  union { unsigned u; float f; } x; x.u = hs << 16;
  return x.f;
}
static __device__ inline v8f v8f_zero() {
  v8f z = {0.f,0.f,0.f,0.f,0.f,0.f,0.f,0.f};
  return z;
}
static __device__ inline v8f wmma_bf16(const V16& a, const V16& b, v8f c) {
  return __builtin_amdgcn_wmma_f32_16x16x32_bf16(
      /*neg_a=*/false, a.v, /*neg_b=*/false, b.v,
      /*c_mod=*/(short)0, c, /*reuse_a=*/false, /*reuse_b=*/false);
}

// ---------------- workspace zeroing ----------------
__global__ void zero_kernel(unsigned* __restrict__ p, long n) {
  long stride = (long)gridDim.x * blockDim.x;
  for (long i = (long)blockIdx.x * blockDim.x + threadIdx.x; i < n; i += stride)
    p[i] = 0u;
}

// ---------------- f32 -> bf16 packed conversion ----------------
__global__ void cvt_kernel(const float* __restrict__ src, unsigned* __restrict__ dst, long n2) {
  long stride = (long)gridDim.x * blockDim.x;
  for (long i = (long)blockIdx.x * blockDim.x + threadIdx.x; i < n2; i += stride) {
    float2 v = ((const float2*)src)[i];
    dst[i] = pack_bf16x2(v.x, v.y);
  }
}

// ---------------- bf16-WMMA GEMM, double-buffered async-to-LDS pipeline ----------------
// Xb: [MPAD, EE] bf16 (pad rows zero), Wb: [EE, EE] bf16 (rows = output cols)
// out = X @ W^T + bias
// mode 0: dst f32 [MROWS, EE]
// mode 1/2: dst bf16 padded Q/K layout [B,H,SPAD,DP]
// mode 3:   dst bf16 transposed V layout [B,H,DP,SPAD]
__global__ __launch_bounds__(128)
void gemm_bf16_kernel(const unsigned short* __restrict__ Xb,
                      const unsigned short* __restrict__ Wb,
                      const float* __restrict__ bias, void* __restrict__ dst,
                      int mode) {
  extern __shared__ unsigned short smem[];   // 16 KB: buf0 {lx@0,lw@4096B}, buf1 {@8192,@12288}

  const int tid  = threadIdx.x;
  const int lane = tid & 31;
  const int wid  = tid >> 5;
  const int wm   = (wid >> 1) * 32;
  const int wn   = (wid & 1)  * 32;
  const int n0   = blockIdx.x * 64;
  const int m0   = blockIdx.y * 64;

  const int mlane = lane & 15;
  const int ka4   = (lane >> 4) * 4;
  const int kb8   = (lane >> 4) * 8;

  // per-thread async-copy assignments: 64 rows x 4 x 16B chunks = 256 chunks per tile
  const int c0 = tid, c1 = 128 + tid;
  const int row0 = c0 >> 2, cg0 = c0 & 3;
  const int row1 = c1 >> 2, cg1 = c1 & 3;
  const unsigned short* xp0 = Xb + (size_t)(m0 + row0) * EE + cg0 * 8;
  const unsigned short* xp1 = Xb + (size_t)(m0 + row1) * EE + cg1 * 8;
  const unsigned short* wp0 = Wb + (size_t)(n0 + row0) * EE + cg0 * 8;
  const unsigned short* wp1 = Wb + (size_t)(n0 + row1) * EE + cg1 * 8;
  const int lxo0 = row0 * 64 + cg0 * 16;         // byte offsets inside a buffer
  const int lxo1 = row1 * 64 + cg1 * 16;

  v8f acc[2][2];
#pragma unroll
  for (int i = 0; i < 2; ++i)
#pragma unroll
    for (int j = 0; j < 2; ++j) acc[i][j] = v8f_zero();

  const int NK = EE / 32;                        // 44 K-chunks

  // issue chunk 0 into buffer 0
  {
    const int base = 0;
    asm volatile("global_load_async_to_lds_b128 %0, %1, off"
                 :: "v"(base + lxo0), "v"(xp0) : "memory");
    asm volatile("global_load_async_to_lds_b128 %0, %1, off"
                 :: "v"(base + lxo1), "v"(xp1) : "memory");
    asm volatile("global_load_async_to_lds_b128 %0, %1, off"
                 :: "v"(base + 4096 + lxo0), "v"(wp0) : "memory");
    asm volatile("global_load_async_to_lds_b128 %0, %1, off"
                 :: "v"(base + 4096 + lxo1), "v"(wp1) : "memory");
  }

  for (int ki = 0; ki < NK; ++ki) {
    const int buf = ki & 1;
    if (ki + 1 < NK) {
      // issue next chunk into the other buffer, then wait for the PREVIOUS 4
      const int kn = (ki + 1) * 32;
      const int base = (buf ^ 1) * 8192;
      asm volatile("global_load_async_to_lds_b128 %0, %1, off"
                   :: "v"(base + lxo0), "v"(xp0 + kn) : "memory");
      asm volatile("global_load_async_to_lds_b128 %0, %1, off"
                   :: "v"(base + lxo1), "v"(xp1 + kn) : "memory");
      asm volatile("global_load_async_to_lds_b128 %0, %1, off"
                   :: "v"(base + 4096 + lxo0), "v"(wp0 + kn) : "memory");
      asm volatile("global_load_async_to_lds_b128 %0, %1, off"
                   :: "v"(base + 4096 + lxo1), "v"(wp1 + kn) : "memory");
      asm volatile("s_wait_asynccnt 0x4" ::: "memory");   // in-order: chunk ki landed
    } else {
      asm volatile("s_wait_asynccnt 0x0" ::: "memory");
    }
    __syncthreads();

    const unsigned* lx32 = (const unsigned*)(smem + buf * 4096);
    const unsigned* lw32 = (const unsigned*)(smem + buf * 4096 + 2048);

    V16 a[2], b[2];
#pragma unroll
    for (int t = 0; t < 2; ++t) {
      int row = wm + t * 16 + mlane;
#pragma unroll
      for (int j = 0; j < 4; ++j) {
        a[t].u[j]     = lx32[row * 16 + ka4 + j];
        a[t].u[4 + j] = lx32[row * 16 + 8 + ka4 + j];
      }
      int ncol = wn + t * 16 + mlane;
#pragma unroll
      for (int j = 0; j < 4; ++j) {
        b[t].u[j]     = lw32[ncol * 16 + ka4 + j];
        b[t].u[4 + j] = lw32[ncol * 16 + 8 + ka4 + j];
      }
    }
#pragma unroll
    for (int i = 0; i < 2; ++i)
#pragma unroll
      for (int j = 0; j < 2; ++j)
        acc[i][j] = wmma_bf16(a[i], b[j], acc[i][j]);

    __syncthreads();   // reads of this buffer done before it is refilled (ki+2)
  }

  // epilogue
  const int rowoff = (lane >> 4) * 8;
#pragma unroll
  for (int ti = 0; ti < 2; ++ti) {
#pragma unroll
    for (int tj = 0; tj < 2; ++tj) {
      int nglob = n0 + wn + tj * 16 + mlane;
      float bv = bias[nglob];
#pragma unroll
      for (int g = 0; g < 8; ++g) {
        int m = m0 + wm + ti * 16 + rowoff + g;
        if (m >= MROWS) continue;
        float v = acc[ti][tj][g] + bv;
        if (mode == 0) {
          ((float*)dst)[(size_t)m * EE + nglob] = v;
        } else {
          int bidx = m / SS, s = m % SS;
          int h = nglob / DH, d = nglob % DH;
          unsigned short hv = f32_to_bf16(v);
          if (mode == 3)
            ((unsigned short*)dst)[((size_t)(bidx * HH + h) * DP + d) * SPAD + s] = hv;
          else
            ((unsigned short*)dst)[((size_t)(bidx * HH + h) * SPAD + s) * DP + d] = hv;
        }
      }
    }
  }
}

// ---------------- RoPE (in-place on padded bf16 Q/K) ----------------
__global__ void rope_kernel(unsigned* __restrict__ Qp, unsigned* __restrict__ Kp,
                            const float* __restrict__ fcos, const float* __restrict__ fsin) {
  int idx = blockIdx.x * blockDim.x + threadIdx.x;
  const int total = BB * HH * SS * (DH / 2);
  if (idx >= total) return;
  int i  = idx % (DH / 2);
  int s  = (idx / (DH / 2)) % SS;
  int bh = idx / ((DH / 2) * SS);
  float c  = fcos[s * (DH / 2) + i];
  float sn = fsin[s * (DH / 2) + i];
  size_t off = ((size_t)bh * SPAD + s) * (DP / 2) + i;
  {
    unsigned q = Qp[off];
    float xr = bf16_to_f32(q & 0xffffu), xi = bf16_to_f32(q >> 16);
    Qp[off] = pack_bf16x2(xr * c - xi * sn, xr * sn + xi * c);
  }
  {
    unsigned k = Kp[off];
    float xr = bf16_to_f32(k & 0xffffu), xi = bf16_to_f32(k >> 16);
    Kp[off] = pack_bf16x2(xr * c - xi * sn, xr * sn + xi * c);
  }
}

// ---------------- flash attention: 1 wave per (b,h, 16-query tile) ----------------
__global__ __launch_bounds__(128)
void attn_kernel(const unsigned short* __restrict__ Qp,
                 const unsigned short* __restrict__ Kp,
                 const unsigned short* __restrict__ Vt,
                 unsigned short* __restrict__ Aob) {
  __shared__ unsigned short pb[4][16 * 32];   // per-wave P transpose tile

  const int lane = threadIdx.x & 31;
  const int widx = threadIdx.x >> 5;
  const int wg   = blockIdx.x * 4 + widx;     // 8*16*66 = 8448 waves total
  const int qt   = wg % (SPAD / 16);
  const int bh   = wg / (SPAD / 16);
  if (bh >= BB * HH) return;                  // wave-uniform guard

  const int mlane  = lane & 15;
  const int ka4    = (lane >> 4) * 4;
  const int kb8    = (lane >> 4) * 8;
  const int rowoff = (lane >> 4) * 8;

  const unsigned* Qw = ((const unsigned*)Qp) + (size_t)bh * SPAD * (DP / 2);
  const unsigned* Kw = ((const unsigned*)Kp) + (size_t)bh * SPAD * (DP / 2);
  const unsigned* Vw = ((const unsigned*)Vt) + (size_t)bh * DP * (SPAD / 2);

  // load Q tile as three 16x32 A-operands (held in registers all loop long)
  V16 qa[3];
  const int srow = qt * 16 + mlane;
#pragma unroll
  for (int ch = 0; ch < 3; ++ch)
#pragma unroll
    for (int j = 0; j < 4; ++j) {
      qa[ch].u[j]     = Qw[(size_t)srow * (DP / 2) + ch * 16 + ka4 + j];
      qa[ch].u[4 + j] = Qw[(size_t)srow * (DP / 2) + ch * 16 + 8 + ka4 + j];
    }

  float mrow[8], lrow[8];
  v8f Ov[6];
#pragma unroll
  for (int g = 0; g < 8; ++g) { mrow[g] = -3.0e38f; lrow[g] = 0.f; }
#pragma unroll
  for (int gi = 0; gi < 6; ++gi) Ov[gi] = v8f_zero();

  const float cl2 = 0.10660035817780522f * 1.4426950408889634f; // scale*log2(e)

  unsigned short* myp = pb[widx];
  const unsigned* myp32 = (const unsigned*)myp;

  for (int kb = 0; kb < SPAD / 32; ++kb) {    // 33 key blocks of 32
    const int k0 = kb * 32;

    // prefetch next block's K and V rows (one row start per lane)
    if (k0 + 32 < SPAD) {
      __builtin_prefetch((const void*)(Kw + (size_t)(k0 + 32 + lane) * (DP / 2)), 0, 3);
      __builtin_prefetch((const void*)(Vw + (size_t)(lane & 15) * (SPAD / 2) + ((k0 + 32) >> 1)), 0, 3);
    }

    // ---- scores: two 16x16 tiles, 3 chained WMMAs each over D=96 ----
    v8f sc[2];
#pragma unroll
    for (int t = 0; t < 2; ++t) {
      sc[t] = v8f_zero();
      int key = k0 + t * 16 + mlane;
#pragma unroll
      for (int ch = 0; ch < 3; ++ch) {
        V16 bb;
#pragma unroll
        for (int j = 0; j < 4; ++j) {
          bb.u[j]     = Kw[(size_t)key * (DP / 2) + ch * 16 + ka4 + j];
          bb.u[4 + j] = Kw[(size_t)key * (DP / 2) + ch * 16 + 8 + ka4 + j];
        }
        sc[t] = wmma_bf16(qa[ch], bb, sc[t]);
      }
    }

    // ---- online softmax update ----
    float alpha[8];
    const int key0 = k0 + mlane;
    const int key1 = k0 + 16 + mlane;
#pragma unroll
    for (int g = 0; g < 8; ++g) {
      float s0 = (key0 < SS) ? sc[0][g] : -3.0e38f;
      float s1 = (key1 < SS) ? sc[1][g] : -3.0e38f;
      float mx = fmaxf(s0, s1);
#pragma unroll
      for (int sh = 1; sh <= 8; sh <<= 1) mx = fmaxf(mx, __shfl_xor(mx, sh, 32));
      float mnew = fmaxf(mrow[g], mx);
      alpha[g] = exp2f((mrow[g] - mnew) * cl2);
      float p0 = exp2f((s0 - mnew) * cl2);
      float p1 = exp2f((s1 - mnew) * cl2);
      float ps = p0 + p1;
#pragma unroll
      for (int sh = 1; sh <= 8; sh <<= 1) ps += __shfl_xor(ps, sh, 32);
      lrow[g] = lrow[g] * alpha[g] + ps;
      mrow[g] = mnew;
      sc[0][g] = p0;
      sc[1][g] = p1;
    }
#pragma unroll
    for (int gi = 0; gi < 6; ++gi)
#pragma unroll
      for (int g = 0; g < 8; ++g) Ov[gi][g] *= alpha[g];

    // ---- transpose P (D-layout -> A-layout) through per-wave LDS ----
    asm volatile("s_wait_dscnt 0" ::: "memory");   // WAR vs last block's reads
#pragma unroll
    for (int t = 0; t < 2; ++t)
#pragma unroll
      for (int g = 0; g < 8; ++g) {
        int row = g + rowoff;
        myp[row * 32 + t * 16 + mlane] = f32_to_bf16(sc[t][g]);
      }
    asm volatile("s_wait_dscnt 0" ::: "memory");   // RAW: stores -> reads

    V16 pa;
#pragma unroll
    for (int j = 0; j < 4; ++j) {
      pa.u[j]     = myp32[mlane * 16 + ka4 + j];
      pa.u[4 + j] = myp32[mlane * 16 + 8 + ka4 + j];
    }

    // ---- O += P @ V  (6 d-column groups) ----
#pragma unroll
    for (int gi = 0; gi < 6; ++gi) {
      V16 vb;
      int d = gi * 16 + mlane;
#pragma unroll
      for (int j = 0; j < 4; ++j) {
        vb.u[j]     = Vw[(size_t)d * (SPAD / 2) + (k0 >> 1) + ka4 + j];
        vb.u[4 + j] = Vw[(size_t)d * (SPAD / 2) + (k0 >> 1) + 8 + ka4 + j];
      }
      Ov[gi] = wmma_bf16(pa, vb, Ov[gi]);
    }
  }

  // ---- epilogue: normalize and scatter to Aob [MPAD, EE] bf16 ----
  const int bidx = bh / HH, h = bh % HH;
#pragma unroll
  for (int gi = 0; gi < 6; ++gi) {
    int d = gi * 16 + mlane;
    if (d >= DH) continue;
#pragma unroll
    for (int g = 0; g < 8; ++g) {
      int s = qt * 16 + rowoff + g;
      if (s >= SS) continue;
      Aob[((size_t)(bidx * SS + s)) * EE + h * DH + d] = f32_to_bf16(Ov[gi][g] / lrow[g]);
    }
  }
}

// ---------------- launch ----------------
extern "C" void kernel_launch(void* const* d_in, const int* in_sizes, int n_in,
                              void* d_out, int out_size, void* d_ws, size_t ws_size,
                              hipStream_t stream) {
  const float* hs  = (const float*)d_in[0];
  const float* fc  = (const float*)d_in[1];
  const float* fs  = (const float*)d_in[2];
  const float* Wq  = (const float*)d_in[3];
  const float* bq  = (const float*)d_in[4];
  const float* Wk  = (const float*)d_in[5];
  const float* bk  = (const float*)d_in[6];
  const float* Wv  = (const float*)d_in[7];
  const float* bv  = (const float*)d_in[8];
  const float* Wo  = (const float*)d_in[9];
  const float* bo  = (const float*)d_in[10];

  char* ws = (char*)d_ws;
  unsigned short* Xb  = (unsigned short*)ws;          // [MPAD, EE] bf16
  unsigned short* Qp  = Xb + XBELEM;                  // [B,H,SPAD,DP] bf16
  unsigned short* Kp  = Qp + PBELEM;
  unsigned short* Vt  = Kp + PBELEM;                  // [B,H,DP,SPAD] bf16
  unsigned short* Aob = Vt + PBELEM;                  // [MPAD, EE] bf16
  unsigned short* Wqb = Aob + XBELEM;                 // [EE,EE] bf16 x4
  unsigned short* Wkb = Wqb + WELEM;
  unsigned short* Wvb = Wkb + WELEM;
  unsigned short* Wob = Wvb + WELEM;

  // zero [Xb | Qp | Kp | Vt | Aob] so padded rows/cols are 0
  long zwords = (long)((2 * XBELEM + 3 * PBELEM) / 2);
  zero_kernel<<<4096, 256, 0, stream>>>((unsigned*)Xb, zwords);

  // f32 -> bf16 conversions
  cvt_kernel<<<4096, 256, 0, stream>>>(hs, (unsigned*)Xb, (long)MROWS * EE / 2);
  cvt_kernel<<<2048, 256, 0, stream>>>(Wq, (unsigned*)Wqb, (long)WELEM / 2);
  cvt_kernel<<<2048, 256, 0, stream>>>(Wk, (unsigned*)Wkb, (long)WELEM / 2);
  cvt_kernel<<<2048, 256, 0, stream>>>(Wv, (unsigned*)Wvb, (long)WELEM / 2);
  cvt_kernel<<<2048, 256, 0, stream>>>(Wo, (unsigned*)Wob, (long)WELEM / 2);

  dim3 ggrid(EE / 64, MPAD / 64);   // 22 x 129
  const size_t lds_bytes = 16384;   // two 8 KB buffers
  gemm_bf16_kernel<<<ggrid, 128, lds_bytes, stream>>>(Xb, Wqb, bq, (void*)Qp, 1);
  gemm_bf16_kernel<<<ggrid, 128, lds_bytes, stream>>>(Xb, Wkb, bk, (void*)Kp, 2);
  gemm_bf16_kernel<<<ggrid, 128, lds_bytes, stream>>>(Xb, Wvb, bv, (void*)Vt, 3);

  int rope_total = BB * HH * SS * (DH / 2);
  rope_kernel<<<(rope_total + 255) / 256, 256, 0, stream>>>((unsigned*)Qp, (unsigned*)Kp, fc, fs);

  int nwaves = BB * HH * (SPAD / 16);       // 8448
  attn_kernel<<<nwaves / 4, 128, 0, stream>>>(Qp, Kp, Vt, Aob);

  gemm_bf16_kernel<<<ggrid, 128, lds_bytes, stream>>>(Aob, Wob, bo, d_out, 0);
}
